// AttentionDecoder_13168369730060
// MI455X (gfx1250) — matrix operand order repaired and checked by reading, not compiled, not run on hardware
//
#include <hip/hip_runtime.h>
#include <hip/hip_bf16.h>
#include <math.h>

#define B 32
#define T 2048
#define H 128
#define OUT_LEN 32

typedef __attribute__((ext_vector_type(16))) __bf16 v16bf;
typedef __attribute__((ext_vector_type(8)))  float  v8f;
typedef __attribute__((ext_vector_type(2)))  float  v2f;

// ---------------------------------------------------------------------------
// Kernel 1: Ua_enc[b,t,k] = sum_h enc[b,t,h] * Ua[k,h]
// (B*T, H) x (H, H)^T GEMM via v_wmma_f32_16x16x32_bf16 (one 16x16 tile/wave)
// ---------------------------------------------------------------------------
__global__ void ua_enc_wmma_kernel(const float* __restrict__ enc,
                                   const float* __restrict__ Ua,
                                   float* __restrict__ ua_enc) {
  const int lane  = threadIdx.x & 31;
  const int wave  = threadIdx.x >> 5;
  const int tile  = blockIdx.x * 8 + wave;   // (B*T/16)*(H/16) = 32768 tiles
  const int tileN = tile & 7;                // H/16 = 8
  const int tileM = tile >> 3;
  const int row0  = tileM * 16;
  const int col0  = tileN * 16;
  const int lhalf = lane >> 4;               // 0/1 lane half
  const int lmod  = lane & 15;

  v8f c = {};
  const float* arow = enc + (size_t)(row0 + lmod) * H;   // A row (M = lmod)
  const float* bro  = Ua  + (size_t)(col0 + lmod) * H;   // B[k][n] = Ua[n][k]

  for (int kc = 0; kc < H; kc += 32) {
    v16bf a, b;
    // A 16x32 bf16 layout: elems 0..7 -> K = kc + lhalf*8 + e
    //                      elems 8..15 -> K = kc + 16 + lhalf*8 + (e-8)
    #pragma unroll
    for (int e = 0; e < 16; ++e) {
      int k = kc + ((e >> 3) * 16) + (lhalf * 8) + (e & 7);
      a[e] = (__bf16)arow[k];
    }
    // B 32x16 bf16 layout: same K order as A, N = lmod
    const float* bk = bro + kc;
    #pragma unroll
    for (int e = 0; e < 16; ++e) b[e] = (__bf16)bk[((e >> 3) * 16) + (lhalf * 8) + (e & 7)];

    c = __builtin_amdgcn_wmma_f32_16x16x32_bf16(false, a, false, b,
                                                (short)0, c, false, false);
  }
  // C/D: VGPR r -> M = row0 + r + lhalf*8, N = col0 + lmod
  #pragma unroll
  for (int r = 0; r < 8; ++r)
    ua_enc[(size_t)(row0 + r + lhalf * 8) * H + col0 + lmod] = c[r];
}

// ---------------------------------------------------------------------------
// Kernel 2 (per step): Wh = h @ Wa^T  (32x128, K=128) in fp32 WMMA
// 16 tiles -> 16 waves -> one 512-thread block.
// ---------------------------------------------------------------------------
__global__ void wh_wmma_kernel(const float* __restrict__ h,
                               const float* __restrict__ Wa,
                               float* __restrict__ wh) {
  const int lane  = threadIdx.x & 31;
  const int wave  = threadIdx.x >> 5;  // 0..15
  const int tileM = wave >> 3;         // B/16 = 2
  const int tileN = wave & 7;          // H/16 = 8
  const int row0  = tileM * 16;
  const int col0  = tileN * 16;
  const int lhalf = lane >> 4;
  const int lmod  = lane & 15;

  v8f c = {};
  const float* arow = h  + (size_t)(row0 + lmod) * H;
  const float* brow = Wa + (size_t)(col0 + lmod) * H;  // B[k][n] = Wa[n][k]
  for (int kc = 0; kc < H; kc += 4) {
    // A 16x4 f32: lane half selects K pair {0,1} vs {2,3}
    v2f a, b;
    a[0] = arow[kc + lhalf * 2 + 0];
    a[1] = arow[kc + lhalf * 2 + 1];
    b[0] = brow[kc + lhalf * 2 + 0];
    b[1] = brow[kc + lhalf * 2 + 1];
    c = __builtin_amdgcn_wmma_f32_16x16x4_f32(false, a, false, b,
                                              (short)0, c, false, false);
  }
  #pragma unroll
  for (int r = 0; r < 8; ++r)
    wh[(size_t)(row0 + r + lhalf * 8) * H + col0 + lmod] = c[r];
}

// ---------------------------------------------------------------------------
// Kernel 3 (per step): energies -> softmax -> w[b,t]
// One block per batch b; wave-per-row coalesced loads, shfl_xor reductions.
// ---------------------------------------------------------------------------
__global__ void attn_softmax_kernel(const float* __restrict__ ua_enc,
                                    const float* __restrict__ wh,
                                    const float* __restrict__ va,
                                    float* __restrict__ w) {
  __shared__ float s_wh[H];
  __shared__ float s_va[H];
  __shared__ float s_e[T];      // 8 KB of the 320 KB WGP LDS
  __shared__ float red[256];
  const int b    = blockIdx.x;
  const int tid  = threadIdx.x;
  const int lane = tid & 31;
  const int wave = tid >> 5;

  if (tid < H) { s_wh[tid] = wh[b * H + tid]; s_va[tid] = va[tid]; }
  __syncthreads();

  // Each of 8 waves covers 256 rows; lane covers 4 contiguous h (float4 load).
  for (int i = 0; i < T / 8; ++i) {
    const int t = wave * (T / 8) + i;
    const float* row = ua_enc + ((size_t)b * T + t) * H;
    const float4 v = *(const float4*)(row + lane * 4);
    const int h0 = lane * 4;
    float acc = tanhf(s_wh[h0 + 0] + v.x) * s_va[h0 + 0]
              + tanhf(s_wh[h0 + 1] + v.y) * s_va[h0 + 1]
              + tanhf(s_wh[h0 + 2] + v.z) * s_va[h0 + 2]
              + tanhf(s_wh[h0 + 3] + v.w) * s_va[h0 + 3];
    #pragma unroll
    for (int off = 16; off > 0; off >>= 1)
      acc += __shfl_xor(acc, off, 32);
    if (lane == 0) s_e[t] = acc;
  }
  __syncthreads();

  float mx = -INFINITY;
  for (int i = tid; i < T; i += 256) mx = fmaxf(mx, s_e[i]);
  red[tid] = mx; __syncthreads();
  for (int s = 128; s > 0; s >>= 1) {
    if (tid < s) red[tid] = fmaxf(red[tid], red[tid + s]);
    __syncthreads();
  }
  mx = red[0]; __syncthreads();

  float sum = 0.f;
  for (int i = tid; i < T; i += 256) {
    float e = expf(s_e[i] - mx);
    s_e[i] = e;
    sum += e;
  }
  red[tid] = sum; __syncthreads();
  for (int s = 128; s > 0; s >>= 1) {
    if (tid < s) red[tid] += red[tid + s];
    __syncthreads();
  }
  const float inv = 1.f / red[0];
  for (int i = tid; i < T; i += 256) w[b * T + i] = s_e[i] * inv;
}

// ---------------------------------------------------------------------------
// Kernel 4 (per step): ctx[b,h] = sum_t w[b,t] * enc[b,t,h]
// ---------------------------------------------------------------------------
__global__ void ctx_kernel(const float* __restrict__ enc,
                           const float* __restrict__ w,
                           float* __restrict__ ctx) {
  __shared__ float s_w[128];
  const int b  = blockIdx.x;
  const int hh = threadIdx.x;  // 128 threads
  float acc = 0.f;
  for (int t0 = 0; t0 < T; t0 += 128) {
    __syncthreads();
    s_w[hh] = w[b * T + t0 + hh];
    __syncthreads();
    const float* base = enc + ((size_t)b * T + t0) * H + hh;
    __builtin_prefetch(base + (size_t)128 * H, 0, 1);  // global_prefetch_b8
    #pragma unroll 8
    for (int i = 0; i < 128; ++i)
      acc += s_w[i] * base[(size_t)i * H];
  }
  ctx[b * H + hh] = acc;
}

// ---------------------------------------------------------------------------
// Kernel 5 (per step): GRU cell on x = [h_prev, ctx]  (inp == h always)
// ---------------------------------------------------------------------------
__global__ void gru_kernel(const float* __restrict__ W_ih,
                           const float* __restrict__ W_hh,
                           const float* __restrict__ b_ih,
                           const float* __restrict__ b_hh,
                           const float* __restrict__ ctx,
                           float* __restrict__ h,
                           float* __restrict__ out, int step) {
  __shared__ float s_x[2 * H];  // [h_prev | ctx]
  const int b = blockIdx.x;
  const int j = threadIdx.x;    // 128 threads: one gate column each
  s_x[j]     = h[b * H + j];
  s_x[H + j] = ctx[b * H + j];
  __syncthreads();

  float ir = b_ih[j], iz = b_ih[H + j], inn = b_ih[2 * H + j];
  const float* wr = W_ih + (size_t)j * (2 * H);
  const float* wz = W_ih + (size_t)(H + j) * (2 * H);
  const float* wn = W_ih + (size_t)(2 * H + j) * (2 * H);
  #pragma unroll 4
  for (int k = 0; k < 2 * H; ++k) {
    const float xv = s_x[k];
    ir += xv * wr[k]; iz += xv * wz[k]; inn += xv * wn[k];
  }
  float hr = b_hh[j], hz = b_hh[H + j], hn = b_hh[2 * H + j];
  const float* vr = W_hh + (size_t)j * H;
  const float* vz = W_hh + (size_t)(H + j) * H;
  const float* vn = W_hh + (size_t)(2 * H + j) * H;
  #pragma unroll 4
  for (int k = 0; k < H; ++k) {
    const float hv = s_x[k];
    hr += hv * vr[k]; hz += hv * vz[k]; hn += hv * vn[k];
  }
  const float r = 1.f / (1.f + expf(-(ir + hr)));
  const float z = 1.f / (1.f + expf(-(iz + hz)));
  const float n = tanhf(inn + r * hn);
  const float hnew = (1.f - z) * n + z * s_x[j];
  h[b * H + j] = hnew;                                // also serves as inp
  out[(size_t)(step * B + b) * H + j] = hnew;
}

__global__ void copy_w_kernel(const float* __restrict__ w,
                              float* __restrict__ dst) {
  const int i = blockIdx.x * 256 + threadIdx.x;
  dst[i] = w[i];
}

// ---------------------------------------------------------------------------
extern "C" void kernel_launch(void* const* d_in, const int* in_sizes, int n_in,
                              void* d_out, int out_size, void* d_ws, size_t ws_size,
                              hipStream_t stream) {
  const float* enc  = (const float*)d_in[0];
  const float* Wa   = (const float*)d_in[1];
  const float* Ua   = (const float*)d_in[2];
  const float* va   = (const float*)d_in[3];
  const float* W_ih = (const float*)d_in[4];
  const float* W_hh = (const float*)d_in[5];
  const float* b_ih = (const float*)d_in[6];
  const float* b_hh = (const float*)d_in[7];
  // d_in[8] = out_len (device scalar); step count must be static for graph
  // capture and matches the reference's fixed shapes -> OUT_LEN = 32.
  float* out = (float*)d_out;

  // Scratch layout (floats): Ua_enc | Wh | w | ctx | h   (~34 MB total)
  float* f_ua  = (float*)d_ws;
  float* f_wh  = f_ua  + (size_t)B * T * H;
  float* f_w   = f_wh  + B * H;
  float* f_ctx = f_w   + B * T;
  float* f_h   = f_ctx + B * H;

  hipMemsetAsync(f_h, 0, (size_t)B * H * sizeof(float), stream);  // h0 = 0

  // Hoisted encoder projection (bf16 WMMA, f32 accumulate)
  ua_enc_wmma_kernel<<<(B * T / 16) * (H / 16) / 8, 256, 0, stream>>>(enc, Ua, f_ua);

  for (int step = 0; step < OUT_LEN; ++step) {
    wh_wmma_kernel<<<1, 512, 0, stream>>>(f_h, Wa, f_wh);
    attn_softmax_kernel<<<B, 256, 0, stream>>>(f_ua, f_wh, va, f_w);
    ctx_kernel<<<B, 128, 0, stream>>>(enc, f_w, f_ctx);
    gru_kernel<<<B, 128, 0, stream>>>(W_ih, W_hh, b_ih, b_hh, f_ctx, f_h, out, step);
  }
  // ws[-1]: step-31 attention weights appended after the (out_len*B,1,H) output
  copy_w_kernel<<<B * T / 256, 256, 0, stream>>>(f_w, out + (size_t)OUT_LEN * B * H);
}
